// ResNetBKLayer_70514773066252
// MI455X (gfx1250) — matrix operand order repaired and checked by reading, not compiled, not run on hardware
//
#include <hip/hip_runtime.h>

// ---------------------------------------------------------------------------
// MI455X (gfx1250) fused implementation.
//
//   Stage A (WMMA): a[row] = (relu(x@W1 + b1) @ W2 + b2) - 2      (h never hits HBM)
//   Stage B (scan): C_i = 1/(a[i-1]-i1 - C_{i-1}),  D_i = 1/(a[i+1]-i1 - D_{i+1})
//                   (contraction => chunked with 64-step halo, fully parallel)
//   Stage C (bw):   G_i = 1/(a[i]-i1 - C_i - D_i);  out = Re(G)*Wp0 + Im(G)*Wp1 + bp
// ---------------------------------------------------------------------------

typedef __attribute__((ext_vector_type(16))) __bf16 v16bf;
typedef __attribute__((ext_vector_type(8)))  float  v8f;

static __device__ __forceinline__ unsigned short f32_to_bf16(float f) {
  unsigned u = __float_as_uint(f);
  u += 0x7FFFu + ((u >> 16) & 1u);          // round-to-nearest-even
  return (unsigned short)(u >> 16);
}
static __device__ __forceinline__ unsigned pack_bf16(float lo, float hi) {
  return (unsigned)f32_to_bf16(lo) | ((unsigned)f32_to_bf16(hi) << 16);
}

// --------------------------------------------------------------------------
// Kernel 0: W1 (f32, [k][n]) -> W1^T (bf16, [n][k]) so B-fragment reads are
// contiguous per lane. 512KB, L2-resident for the whole GEMM.
// --------------------------------------------------------------------------
__global__ __launch_bounds__(256) void cvt_w1t_kernel(
    const float* __restrict__ W1, unsigned short* __restrict__ w1t) {
  int idx = blockIdx.x * 256 + threadIdx.x;   // 0 .. 512*512-1, idx = n*512 + k
  int n = idx >> 9, k = idx & 511;
  w1t[idx] = f32_to_bf16(W1[k * 512 + n]);
}

// --------------------------------------------------------------------------
// Kernel 1: fused  relu(x@W1+b1)@W2+b2-2  -> a[32768]
// 256 blocks x 256 threads (8 waves). Each block: 128 rows. Each wave: 16 rows.
// x tile staged in LDS as bf16 with 520-element row stride (1040B = 260 dwords,
// 260 mod 64 = 4 -> the 16 row-readers hit distinct banks).
// --------------------------------------------------------------------------
__global__ __launch_bounds__(256) void fused_mlp_kernel(
    const float* __restrict__ x, const unsigned short* __restrict__ w1t,
    const float* __restrict__ b1, const float* __restrict__ W2,
    const float* __restrict__ b2, float* __restrict__ a_out) {
  __shared__ unsigned short lA[128 * 520];     // 130 KB of the 320 KB WGP LDS
  const int tid = threadIdx.x;

  // ---- stage x tile: 2 threads per row, f32 -> bf16, 16B LDS stores --------
  {
    const int row = tid >> 1, half = tid & 1;
    const float* src = x + (size_t)(blockIdx.x * 128 + row) * 512 + half * 256;
    unsigned short* dst = &lA[row * 520 + half * 256];
#pragma unroll 4
    for (int j = 0; j < 32; ++j) {
      float4 f0 = ((const float4*)src)[2 * j];
      float4 f1 = ((const float4*)src)[2 * j + 1];
      uint4 p;
      p.x = pack_bf16(f0.x, f0.y); p.y = pack_bf16(f0.z, f0.w);
      p.z = pack_bf16(f1.x, f1.y); p.w = pack_bf16(f1.z, f1.w);
      *(uint4*)(dst + 8 * j) = p;
    }
  }
  __syncthreads();

  const int wave = tid >> 5, lane = tid & 31;
  const int l15 = lane & 15, hi = lane >> 4;
  const int wr0 = wave * 16;
  // A-fragment (ISA 16-bit 16x32): lanes 0-15 row M=l, K {0..7,16..23};
  // lanes 16-31 same rows, K {8..15,24..31}  -> two ds_load_b128 per k-step.
  const unsigned short* arow = &lA[(wr0 + l15) * 520 + hi * 8];

  union U16 { uint4 u[2]; v16bf v; };
  float vp[8];
#pragma unroll
  for (int r = 0; r < 8; ++r) vp[r] = 0.f;

  for (int nt = 0; nt < 32; ++nt) {            // 32 N-tiles of 16 cols
    v8f acc = {0.f, 0.f, 0.f, 0.f, 0.f, 0.f, 0.f, 0.f};
    // B-fragment (32x16): lane -> col N = nt*16 + l15; lane-half selects
    // K {0..7,16..23} / {8..15,24..31} (same as A) -> two 16B reads per k-step.
    const unsigned short* brow = w1t + (size_t)(nt * 16 + l15) * 512 + hi * 8;
#pragma unroll
    for (int kk = 0; kk < 16; ++kk) {          // K = 512 in steps of 32
      U16 fa, fb;
      fa.u[0] = *(const uint4*)(arow + kk * 32);
      fa.u[1] = *(const uint4*)(arow + kk * 32 + 16);
      fb.u[0] = *(const uint4*)(brow + kk * 32);
      fb.u[1] = *(const uint4*)(brow + kk * 32 + 16);
      acc = __builtin_amdgcn_wmma_f32_16x16x32_bf16(
          false, fa.v, false, fb.v, (short)0, acc, false, false);
    }
    const float b1v = b1[nt * 16 + l15];
    const float w2v = W2[nt * 16 + l15];
#pragma unroll
    for (int r = 0; r < 8; ++r) {              // relu + fold into v partials
      float h = acc[r] + b1v;
      h = h > 0.f ? h : 0.f;
      vp[r] += h * w2v;
    }
  }
  // C/D layout: VGPR r holds (M=r, lanes 0-15) and (M=r+8, lanes 16-31).
  // Reduce the 16-lane groups, then lanes 0 and 16 own rows wr0+r / wr0+8+r.
#pragma unroll
  for (int r = 0; r < 8; ++r) {
    float v = vp[r];
    v += __shfl_xor(v, 1); v += __shfl_xor(v, 2);
    v += __shfl_xor(v, 4); v += __shfl_xor(v, 8);
    vp[r] = v;
  }
  if (l15 == 0) {
    const float b2v = b2[0];
    const int g = blockIdx.x * 128 + wr0 + hi * 8;
#pragma unroll
    for (int r = 0; r < 8; ++r) a_out[g + r] = vp[r] + b2v - 2.0f;
  }
}

// --------------------------------------------------------------------------
// Kernel 2: halo'd parallel continued fractions. 256 threads = 8 batches x
// 16 chunks x 2 directions. Each chunk of 256 runs a 64-step halo first;
// the recurrence contracts ~4x/step so the boundary error is far below ulp.
// --------------------------------------------------------------------------
__global__ __launch_bounds__(256) void tridiag_scan_kernel(
    const float* __restrict__ a, float2* __restrict__ c, float2* __restrict__ d) {
  const int tid = threadIdx.x;
  const int dir = tid >> 7, rem = tid & 127;
  const int b = rem >> 4, ch = rem & 15;
  const float* ar = a + b * 4096;
  if (dir == 0) {                       // forward: C_{i+1} = 1/(a[i]-i - C_i)
    float2* crow = c + b * 4096;
    const int s = ch * 256, e = s + 256;
    int i = (s == 0) ? 0 : s - 64;
    float cr = 0.f, ci = 0.f;
    for (; i < e; ++i) {
      if (i >= s) crow[i] = make_float2(cr, ci);
      float xr = ar[i] - cr, xi = -1.f - ci;
      float inv = 1.f / (xr * xr + xi * xi);
      cr = xr * inv; ci = -xi * inv;
    }
  } else {                              // backward: D_{i-1} = 1/(a[i]-i - D_i)
    float2* drow = d + b * 4096;
    const int s = ch * 256, e = s + 255;
    int i = (e == 4095) ? 4095 : e + 64;
    float dr = 0.f, di = 0.f;
    for (; i >= s; --i) {
      if (i <= e) drow[i] = make_float2(dr, di);
      float xr = ar[i] - dr, xi = -1.f - di;
      float inv = 1.f / (xr * xr + xi * xi);
      dr = xr * inv; di = -xi * inv;
    }
  }
}

// --------------------------------------------------------------------------
// Kernel 3: out[row][:] = Re(G)*Wp[0][:] + Im(G)*Wp[1][:] + bp.  Pure
// bandwidth (64 MB out); float4 stores, 2 rows per 256-thread block.
// --------------------------------------------------------------------------
__global__ __launch_bounds__(256) void expand_out_kernel(
    const float* __restrict__ a, const float2* __restrict__ c,
    const float2* __restrict__ d, const float* __restrict__ Wp,
    const float* __restrict__ bp, float* __restrict__ out) {
  const int tid = threadIdx.x;
  const int row = blockIdx.x * 2 + (tid >> 7);
  const int col = (tid & 127) * 4;
  const float ar = a[row];
  const float2 cc = c[row], dd = d[row];
  float xr = ar - cc.x - dd.x, xi = -1.f - cc.y - dd.y;
  float inv = 1.f / (xr * xr + xi * xi);
  float gr = xr * inv, gi = -xi * inv;
  const float4 w0 = *(const float4*)(Wp + col);
  const float4 w1 = *(const float4*)(Wp + 512 + col);
  const float4 bv = *(const float4*)(bp + col);
  float4 o;
  o.x = gr * w0.x + gi * w1.x + bv.x;
  o.y = gr * w0.y + gi * w1.y + bv.y;
  o.z = gr * w0.z + gi * w1.z + bv.z;
  o.w = gr * w0.w + gi * w1.w + bv.w;
  *(float4*)(out + (size_t)row * 512 + col) = o;
}

// --------------------------------------------------------------------------
extern "C" void kernel_launch(void* const* d_in, const int* in_sizes, int n_in,
                              void* d_out, int out_size, void* d_ws, size_t ws_size,
                              hipStream_t stream) {
  (void)in_sizes; (void)n_in; (void)out_size; (void)ws_size;
  const float* x  = (const float*)d_in[0];
  const float* W1 = (const float*)d_in[1];
  const float* b1 = (const float*)d_in[2];
  const float* W2 = (const float*)d_in[3];
  const float* b2 = (const float*)d_in[4];
  const float* Wp = (const float*)d_in[5];
  const float* bp = (const float*)d_in[6];
  float* out = (float*)d_out;

  char* ws = (char*)d_ws;                                   // layout:
  unsigned short* w1t = (unsigned short*)(ws);              // 512 KB  W1^T bf16
  float*  a  = (float*)(ws + 524288);                       // 128 KB  he_diag
  float2* c  = (float2*)(ws + 655360);                      // 256 KB  fwd CF
  float2* dd = (float2*)(ws + 917504);                      // 256 KB  bwd CF

  cvt_w1t_kernel   <<<1024, 256, 0, stream>>>(W1, w1t);
  fused_mlp_kernel <<<256,  256, 0, stream>>>(x, w1t, b1, W2, b2, a);
  tridiag_scan_kernel<<<1,  256, 0, stream>>>(a, c, dd);
  expand_out_kernel<<<16384, 256, 0, stream>>>(a, c, dd, Wp, bp, out);
}
